// GPT2Attention_5463198400851
// MI455X (gfx1250) — matrix operand order repaired and checked by reading, not compiled, not run on hardware
//
#include <hip/hip_runtime.h>
#include <stdint.h>

// ---------------- problem constants ----------------
#define BB   2
#define SS   2048
#define DD   1024
#define HH   16
#define DHH  64
#define MQ   (BB*SS)          // 4096 rows of x / ctx
#define N_QKV (3*DD)          // 3072
#define KD   DD               // 1024 reduction dim for both GEMMs

typedef __attribute__((ext_vector_type(16))) __bf16 v16bf;
typedef __attribute__((ext_vector_type(8)))  float  v8f;

struct alignas(16) I4 { int x, y, z, w; };
union FragU { struct P { I4 lo, hi; } p; v16bf v; };

// round-to-nearest-even fp32 -> bf16
__device__ __forceinline__ uint16_t f2bf(float f) {
  uint32_t u = __float_as_uint(f);
  uint32_t r = u + 0x7FFFu + ((u >> 16) & 1u);
  if ((u & 0x7F800000u) == 0x7F800000u) r = u;   // Inf/NaN passthrough
  return (uint16_t)(r >> 16);
}

// A-matrix 16x32 bf16 fragment (ISA 7.12.2): lanes 0-15 hold K 0-7 & 16-23,
// lanes 16-31 hold K 8-15 & 24-31.  Two 16B loads, 32B apart.
__device__ __forceinline__ v16bf ldA(const uint16_t* p, long rowOffBytes, int k0, int lane) {
  const char* a = (const char*)p + rowOffBytes + (long)k0 * 2 + ((lane & 16) ? 16 : 0);
  FragU u;
  u.p.lo = *(const I4*)(a);
  u.p.hi = *(const I4*)(a + 32);
  return u.v;
}

// B-matrix 32x16 bf16 fragment: lane = column; lanes 0-15 hold K 0-7 & 16-23,
// lanes 16-31 hold K 8-15 & 24-31.  Two 16B loads, 32B apart.
__device__ __forceinline__ v16bf ldB(const uint16_t* p, long rowOffBytes, int k0, int lane) {
  const char* a = (const char*)p + rowOffBytes + (long)k0 * 2 + ((lane & 16) ? 16 : 0);
  FragU u;
  u.p.lo = *(const I4*)(a);
  u.p.hi = *(const I4*)(a + 32);
  return u.v;
}

#define WMMA_BF16(A_, B_, C_) \
  __builtin_amdgcn_wmma_f32_16x16x32_bf16(false, (A_), false, (B_), (short)0, (C_), false, false)

// ---------------- conversion kernels ----------------
__global__ void __launch_bounds__(256) cvt_f32_bf16(const float* __restrict__ src,
                                                    uint16_t* __restrict__ dst, int n) {
  int i = blockIdx.x * 256 + threadIdx.x;
  if (i < n) dst[i] = f2bf(src[i]);
}

// w[K][N] fp32 -> wT[N][K] bf16
__global__ void __launch_bounds__(256) transcvt(const float* __restrict__ w,
                                                uint16_t* __restrict__ wT, int K, int N) {
  int i = blockIdx.x * 256 + threadIdx.x;
  if (i < K * N) {
    int k = i / N, n = i - k * N;
    wT[(long)n * K + k] = f2bf(w[i]);
  }
}

// ---------------- QKV GEMM ----------------
// A: x_bf16 [4096][1024]; BT: w_attnT [3072][1024]; out scattered to qkv[3][B][H][S][64] bf16
__global__ void __launch_bounds__(256) qkv_gemm(const uint16_t* __restrict__ A,
                                                const uint16_t* __restrict__ BT,
                                                const float* __restrict__ bias,
                                                uint16_t* __restrict__ qkv) {
  const int tid = threadIdx.x, lane = tid & 31, wave = tid >> 5;
  const int mlane = lane & 15, hsel = lane >> 4;
  const int bm = blockIdx.x & 63;          // 4096/64
  const int bn = blockIdx.x >> 6;          // 3072/128
  const int wm = wave & 1, wn = wave >> 1; // 2x4 wave grid, 32x32 per wave
  const int rowA0 = bm * 64 + wm * 32;
  const int colB0 = bn * 128 + wn * 32;

  v8f acc[2][2];
  v8f z = {0.f,0.f,0.f,0.f,0.f,0.f,0.f,0.f};
  acc[0][0] = z; acc[0][1] = z; acc[1][0] = z; acc[1][1] = z;

  const long aRow0 = (long)(rowA0 + mlane)      * (KD * 2);
  const long aRow1 = (long)(rowA0 + 16 + mlane) * (KD * 2);
  const long bRow0 = (long)(colB0 + mlane)      * (KD * 2);
  const long bRow1 = (long)(colB0 + 16 + mlane) * (KD * 2);

  for (int k0 = 0; k0 < KD; k0 += 32) {
    v16bf a0 = ldA(A, aRow0, k0, lane);
    v16bf a1 = ldA(A, aRow1, k0, lane);
    v16bf b0 = ldB(BT, bRow0, k0, lane);
    v16bf b1 = ldB(BT, bRow1, k0, lane);
    acc[0][0] = WMMA_BF16(a0, b0, acc[0][0]);
    acc[0][1] = WMMA_BF16(a0, b1, acc[0][1]);
    acc[1][0] = WMMA_BF16(a1, b0, acc[1][0]);
    acc[1][1] = WMMA_BF16(a1, b1, acc[1][1]);
  }

  // epilogue: +bias, scatter into qkv[3][B][H][S][64] as bf16
  for (int i = 0; i < 2; ++i)
    for (int j = 0; j < 2; ++j) {
      int n = colB0 + j * 16 + mlane;
      float bv = bias[n];
      int sel = n >> 10, d = n & 1023, h = d >> 6, dh = d & 63;
      for (int r = 0; r < 8; ++r) {
        int m = rowA0 + i * 16 + r + hsel * 8;     // C layout: VGPR r -> M = r + 8*(lane>>4)
        int b = m >> 11, s = m & 2047;
        float val = acc[i][j][r] + bv;
        qkv[(((((long)sel * BB + b) * HH + h) * SS) + s) * DHH + dh] = f2bf(val);
      }
    }
}

// ---------------- flash attention ----------------
// qkv[3][B][H][S][64] bf16 -> ctx[B][S][D] bf16 (heads merged)
__global__ void __launch_bounds__(128) flash_attn(const uint16_t* __restrict__ qkv,
                                                  uint16_t* __restrict__ ctx) {
  const long plane = (long)BB * HH * SS * DHH;
  const int bx = blockIdx.x;
  const int qt = bx & 31;            // 32 q-tiles of 64 rows
  const int h  = (bx >> 5) & 15;
  const int b  = bx >> 9;
  const long bh = ((long)(b * HH + h)) * SS * DHH;
  const uint16_t* Qp = qkv + bh;
  const uint16_t* Kp = qkv + plane + bh;
  const uint16_t* Vp = qkv + 2 * plane + bh;

  __shared__ __align__(16) float    Sld[64][65];
  __shared__ __align__(16) uint16_t Pld[64][72];   // 144B row stride (16B aligned)
  __shared__ __align__(16) uint16_t Vt [64][72];   // V transposed: [dh][kv]
  __shared__ float alpha_s[64];
  __shared__ float lsum_s[64];

  const int tid = threadIdx.x, lane = tid & 31, wave = tid >> 5;
  const int mlane = lane & 15, hsel = lane >> 4;

  // Q fragments for this wave's 16 rows (loaded once; DH=64 -> 2 K-steps)
  const long qRow = (long)(qt * 64 + wave * 16 + mlane) * (DHH * 2);
  v16bf qa0 = ldA(Qp, qRow, 0, lane);
  v16bf qa1 = ldA(Qp, qRow, 32, lane);

  v8f z = {0.f,0.f,0.f,0.f,0.f,0.f,0.f,0.f};
  v8f oacc[4]; oacc[0] = z; oacc[1] = z; oacc[2] = z; oacc[3] = z;
  float m_i = -1e30f, l_i = 0.f;     // live in threads 0..63 (row = tid)

  for (int c = 0; c <= qt; ++c) {
    // stage V chunk transposed: Vt[dh][kv_local]
    for (int e = tid; e < 64 * 64; e += 128) {
      int si = e >> 6, dh = e & 63;
      Vt[dh][si] = Vp[(long)(c * 64 + si) * DHH + dh];
    }
    // scores S = Q K^T (fragments straight from L2-resident global)
    v8f sacc[4]; sacc[0] = z; sacc[1] = z; sacc[2] = z; sacc[3] = z;
    for (int nt = 0; nt < 4; ++nt) {
      long kRow = (long)(c * 64 + nt * 16 + mlane) * (DHH * 2);
      v16bf kb0 = ldB(Kp, kRow, 0, lane);
      sacc[nt] = WMMA_BF16(qa0, kb0, sacc[nt]);
      v16bf kb1 = ldB(Kp, kRow, 32, lane);
      sacc[nt] = WMMA_BF16(qa1, kb1, sacc[nt]);
    }
    for (int nt = 0; nt < 4; ++nt)
      for (int r = 0; r < 8; ++r)
        Sld[wave * 16 + r + hsel * 8][nt * 16 + mlane] = sacc[nt][r];
    __syncthreads();

    // per-row online softmax (thread tid owns row tid); mask value -10000 like reference
    if (tid < 64) {
      int row = tid, qg = qt * 64 + row;
      float mx = -1e30f;
      for (int j = 0; j < 64; ++j) {
        int kg = c * 64 + j;
        float s = (kg <= qg) ? Sld[row][j] * 0.125f : -10000.0f;
        Sld[row][j] = s;
        mx = fmaxf(mx, s);
      }
      float mnew = fmaxf(m_i, mx);
      float al = __expf(m_i - mnew);
      float ls = 0.f;
      for (int j = 0; j < 64; ++j) {
        float e = __expf(Sld[row][j] - mnew);
        ls += e;
        Pld[row][j] = f2bf(e);
      }
      m_i = mnew;
      l_i = l_i * al + ls;
      alpha_s[row] = al;
    }
    __syncthreads();

    // rescale running output, then O += P V
    for (int nt = 0; nt < 4; ++nt)
      for (int r = 0; r < 8; ++r)
        oacc[nt][r] *= alpha_s[wave * 16 + hsel * 8 + r];
    for (int ks = 0; ks < 2; ++ks) {
      v16bf pa = ldA(&Pld[0][0], (long)(wave * 16 + mlane) * 144, ks * 32, lane);
      for (int nt = 0; nt < 4; ++nt) {
        v16bf vb = ldB(&Vt[0][0], (long)(nt * 16 + mlane) * 144, ks * 32, lane);
        oacc[nt] = WMMA_BF16(pa, vb, oacc[nt]);
      }
    }
    __syncthreads();
  }

  if (tid < 64) lsum_s[tid] = l_i;
  __syncthreads();

  for (int nt = 0; nt < 4; ++nt)
    for (int r = 0; r < 8; ++r) {
      int rloc = wave * 16 + hsel * 8 + r;
      int srow = qt * 64 + rloc;
      float v = oacc[nt][r] / lsum_s[rloc];
      int dh = nt * 16 + mlane;
      ctx[((long)(b * SS + srow)) * DD + h * DHH + dh] = f2bf(v);
    }
}

// ---------------- output projection ----------------
// A: ctx bf16 [4096][1024]; BT: w_projT bf16 [1024][1024]; out fp32 [4096][1024]
__global__ void __launch_bounds__(256) proj_gemm(const uint16_t* __restrict__ A,
                                                 const uint16_t* __restrict__ BT,
                                                 const float* __restrict__ bias,
                                                 float* __restrict__ out) {
  const int tid = threadIdx.x, lane = tid & 31, wave = tid >> 5;
  const int mlane = lane & 15, hsel = lane >> 4;
  const int bm = blockIdx.x & 63;          // 4096/64
  const int bn = blockIdx.x >> 6;          // 1024/128
  const int wm = wave & 1, wn = wave >> 1;
  const int rowA0 = bm * 64 + wm * 32;
  const int colB0 = bn * 128 + wn * 32;

  v8f acc[2][2];
  v8f z = {0.f,0.f,0.f,0.f,0.f,0.f,0.f,0.f};
  acc[0][0] = z; acc[0][1] = z; acc[1][0] = z; acc[1][1] = z;

  const long aRow0 = (long)(rowA0 + mlane)      * (KD * 2);
  const long aRow1 = (long)(rowA0 + 16 + mlane) * (KD * 2);
  const long bRow0 = (long)(colB0 + mlane)      * (KD * 2);
  const long bRow1 = (long)(colB0 + 16 + mlane) * (KD * 2);

  for (int k0 = 0; k0 < KD; k0 += 32) {
    v16bf a0 = ldA(A, aRow0, k0, lane);
    v16bf a1 = ldA(A, aRow1, k0, lane);
    v16bf b0 = ldB(BT, bRow0, k0, lane);
    v16bf b1 = ldB(BT, bRow1, k0, lane);
    acc[0][0] = WMMA_BF16(a0, b0, acc[0][0]);
    acc[0][1] = WMMA_BF16(a0, b1, acc[0][1]);
    acc[1][0] = WMMA_BF16(a1, b0, acc[1][0]);
    acc[1][1] = WMMA_BF16(a1, b1, acc[1][1]);
  }

  for (int i = 0; i < 2; ++i)
    for (int j = 0; j < 2; ++j) {
      int n = colB0 + j * 16 + mlane;
      float bv = bias[n];
      for (int r = 0; r < 8; ++r) {
        int m = rowA0 + i * 16 + r + hsel * 8;
        out[(long)m * DD + n] = acc[i][j][r] + bv;
      }
    }
}

// ---------------- launch ----------------
extern "C" void kernel_launch(void* const* d_in, const int* in_sizes, int n_in,
                              void* d_out, int out_size, void* d_ws, size_t ws_size,
                              hipStream_t stream) {
  const float* x      = (const float*)d_in[0];
  const float* w_attn = (const float*)d_in[1];
  const float* b_attn = (const float*)d_in[2];
  const float* w_proj = (const float*)d_in[3];
  const float* b_proj = (const float*)d_in[4];
  float* out = (float*)d_out;

  char* ws = (char*)d_ws;
  uint16_t* xb  = (uint16_t*)(ws + 0);                       //  8 MiB: x bf16
  uint16_t* waT = (uint16_t*)(ws + (size_t)8  * (1 << 20));  //  6 MiB: w_attn^T bf16
  uint16_t* wpT = (uint16_t*)(ws + (size_t)14 * (1 << 20));  //  2 MiB: w_proj^T bf16
  uint16_t* qkv = (uint16_t*)(ws + (size_t)16 * (1 << 20));  // 24 MiB: q/k/v bf16
  uint16_t* ctx = (uint16_t*)(ws + (size_t)40 * (1 << 20));  //  8 MiB: ctx bf16

  { int n = MQ * DD;                                   // x -> bf16
    cvt_f32_bf16<<<(n + 255) / 256, 256, 0, stream>>>(x, xb, n); }
  { int n = KD * N_QKV;                                // w_attn -> transposed bf16
    transcvt<<<(n + 255) / 256, 256, 0, stream>>>(w_attn, waT, KD, N_QKV); }
  { int n = KD * DD;                                   // w_proj -> transposed bf16
    transcvt<<<(n + 255) / 256, 256, 0, stream>>>(w_proj, wpT, KD, DD); }

  qkv_gemm<<<64 * (N_QKV / 128), 256, 0, stream>>>(xb, waT, b_attn, qkv);
  flash_attn<<<BB * HH * (SS / 64), 128, 0, stream>>>(qkv, ctx);
  proj_gemm<<<64 * (DD / 128), 256, 0, stream>>>(ctx, wpT, b_proj, out);
}
